// MultiHeadSelfAttention_14121852469409
// MI455X (gfx1250) — matrix operand order repaired and checked by reading, not compiled, not run on hardware
//
#include <hip/hip_runtime.h>
#include <stdint.h>

// Problem constants (match reference)
#define SEQ    2048
#define DMODEL 512
#define NH     8
#define DH     512                 // per-head key/value dim
#define BATCH  2
#define MROWS  (BATCH * SEQ)       // 4096
#define NCOLS  (NH * DH)           // 4096

#define PSTRIDE 2056               // padded P row stride (bf16 elems)
#define KSTRIDE 520                // padded K-stage row stride (bf16 elems)
#define BSTRIDE 72                 // padded B-stage row stride (bf16 elems) = 144B
#define KC      64                 // k-depth per GEMM stage (2 WMMA chunks)

typedef __attribute__((ext_vector_type(16))) __bf16 v16bf;
typedef __attribute__((ext_vector_type(8)))  __bf16 v8bf;
typedef __attribute__((ext_vector_type(8)))  float  v8f;

union ABf { v16bf v; v8bf h[2]; };
struct ASet { ABf a[2][2]; };      // [m-tile][k-chunk]

__device__ __forceinline__ unsigned short f2bf(float f) {
  union { float f; unsigned int u; } c; c.f = f;
  unsigned int u = c.u;
  unsigned int r = u + 0x7FFFu + ((u >> 16) & 1u);   // round-to-nearest-even
  return (unsigned short)(r >> 16);
}
__device__ __forceinline__ float bf2f(unsigned short h) {
  union { unsigned int u; float f; } c; c.u = ((unsigned int)h) << 16;
  return c.f;
}

// --- CDNA5 async copy: global -> LDS, tracked by ASYNCcnt --------------------
__device__ __forceinline__ void async_b128(unsigned lds_off, const void* g) {
  asm volatile("global_load_async_to_lds_b128 %0, %1, off"
               :: "v"(lds_off), "v"(g) : "memory");
}
#define WAIT_ASYNC_LE(n) asm volatile("s_wait_asynccnt " #n ::: "memory")
#define LDS_OFF(p) ((unsigned)(uintptr_t)(p))

// ---------------------------------------------------------------------------
// Kernel 1: x + sinusoidal PE, LayerNorm, emit bf16 activations [MROWS][DMODEL]
// ---------------------------------------------------------------------------
__global__ __launch_bounds__(256) void prep_kernel(const float* __restrict__ x,
                                                   const float* __restrict__ gamma,
                                                   const float* __restrict__ beta,
                                                   unsigned short* __restrict__ xn) {
  __shared__ float red[256];
  const int row = blockIdx.x;            // b*SEQ + s
  const int s   = row & (SEQ - 1);
  const int tid = threadIdx.x;
  const int i0  = tid * 2;
  const float* xr = x + (size_t)row * DMODEL;

  const float rate = __expf(-0.0359778925f * (float)tid);  // 2*ln(1e4)/512
  const float ang  = (float)s * rate;
  float v0 = xr[i0]     + __sinf(ang);
  float v1 = xr[i0 + 1] + __cosf(ang);

  red[tid] = v0 + v1;
  __syncthreads();
  for (int off = 128; off > 0; off >>= 1) {
    if (tid < off) red[tid] += red[tid + off];
    __syncthreads();
  }
  const float mu = red[0] * (1.0f / DMODEL);
  __syncthreads();

  float d0 = v0 - mu, d1 = v1 - mu;
  red[tid] = d0 * d0 + d1 * d1;
  __syncthreads();
  for (int off = 128; off > 0; off >>= 1) {
    if (tid < off) red[tid] += red[tid + off];
    __syncthreads();
  }
  const float rstd = rsqrtf(red[0] * (1.0f / DMODEL) + 1e-6f);

  const float y0 = d0 * rstd * gamma[i0]     + beta[i0];
  const float y1 = d1 * rstd * gamma[i0 + 1] + beta[i0 + 1];
  xn[(size_t)row * DMODEL + i0]     = f2bf(y0);
  xn[(size_t)row * DMODEL + i0 + 1] = f2bf(y1);
}

// ---------------------------------------------------------------------------
// Kernel 2: fp32 weight [K][N] -> bf16 transposed [N][K] (LDS-tiled 32x32)
// ---------------------------------------------------------------------------
__global__ __launch_bounds__(256) void transpose_bf16(const float* __restrict__ W,
                                                      unsigned short* __restrict__ WT,
                                                      int rows /*K*/, int cols /*N*/) {
  __shared__ unsigned short tile[32][33];
  const int bn = blockIdx.x * 32;
  const int bk = blockIdx.y * 32;
  const int tx = threadIdx.x & 31;
  const int ty = threadIdx.x >> 5;
  for (int r = ty; r < 32; r += 8)
    tile[r][tx] = f2bf(W[(size_t)(bk + r) * cols + bn + tx]);
  __syncthreads();
  for (int r = ty; r < 32; r += 8)
    WT[(size_t)(bn + r) * rows + bk + tx] = tile[tx][r];
}

// ---------------------------------------------------------------------------
// Kernel 3/5: bf16 GEMM  C[M][N] = A[M][K] * BT[N][K]^T + bias
// Block = 4 waves; each wave owns a 32x64 tile (2 m-tiles x 4 n-tiles,
// 8 accumulators). The shared 64-row B panel is staged 64-k-deep per stage
// into double-buffered LDS with async copies; 16 WMMAs per barrier pair.
// Manual 2x unroll ping-pongs buffers and A register sets (no copies).
// B operands of a k-chunk are batch-loaded before the chunk's 8 WMMAs so
// the compiler can issue partial s_wait_dscnt and overlap LDS reads w/ math.
// ---------------------------------------------------------------------------
__global__ __launch_bounds__(128) void gemm_bf16(const unsigned short* __restrict__ A,
                                                 const unsigned short* __restrict__ BT,
                                                 const float* __restrict__ bias,
                                                 float* __restrict__ outF,
                                                 unsigned short* __restrict__ outB,
                                                 int Kdim, int mode) {
  __shared__ unsigned short Bst[2][64 * BSTRIDE];   // 2 x 9216B

  const int tid  = threadIdx.x;
  const int wave = tid >> 5;
  const int lane = tid & 31;
  const int lm   = lane & 15;
  const int half = (lane >> 4) & 1;
  const int nb   = blockIdx.x * 64;
  const int mb   = blockIdx.y * 128 + wave * 32;

  const unsigned short* arow0 = A + (size_t)(mb + lm) * Kdim;
  const unsigned short* arow1 = arow0 + (size_t)16 * Kdim;
  const unsigned short* bpan  = BT + (size_t)nb * Kdim;
  const unsigned bstBase = LDS_OFF(&Bst[0][0]);
  const int ns = Kdim / KC;                          // stages (8 or 64, even)

  v8f acc[2][4];
  #pragma unroll
  for (int mt = 0; mt < 2; ++mt)
    #pragma unroll
    for (int t = 0; t < 4; ++t) acc[mt][t] = (v8f){};

  // stage B k-slab s (64 rows x 128B) into buffer buf: 4 async b128 / thread
  auto stageB = [&](int s, int buf) {
    const unsigned base = bstBase + (unsigned)buf * (64 * BSTRIDE * 2);
    #pragma unroll
    for (int i = 0; i < 4; ++i) {
      const int seg = tid + i * 128;            // 0..511 16B segments
      const int row = seg >> 3, part = seg & 7;
      const char* src = (const char*)(bpan + (size_t)row * Kdim + s * KC) + part * 16;
      async_b128(base + (unsigned)(row * (BSTRIDE * 2) + part * 16), src);
    }
  };
  auto loadA = [&](int s, ASet& st) {
    #pragma unroll
    for (int mt = 0; mt < 2; ++mt) {
      const unsigned short* ar = (mt ? arow1 : arow0) + s * KC;
      #pragma unroll
      for (int c = 0; c < 2; ++c) {
        st.a[mt][c].h[0] = *(const v8bf*)(ar + c * 32 + half * 16);
        st.a[mt][c].h[1] = *(const v8bf*)(ar + c * 32 + half * 16 + 8);
      }
    }
  };
  auto compute = [&](int buf, ASet& st) {
    const unsigned short* bb = &Bst[buf][lm * BSTRIDE + half * 16];
    #pragma unroll
    for (int c = 0; c < 2; ++c) {
      v16bf b[4];
      #pragma unroll
      for (int t = 0; t < 4; ++t)
        b[t] = *(const v16bf*)(bb + t * 16 * BSTRIDE + c * 32);
      #pragma unroll
      for (int t = 0; t < 4; ++t) {
        acc[0][t] = __builtin_amdgcn_wmma_f32_16x16x32_bf16(false, st.a[0][c].v, false, b[t],
                                                            (short)0, acc[0][t], false, false);
        acc[1][t] = __builtin_amdgcn_wmma_f32_16x16x32_bf16(false, st.a[1][c].v, false, b[t],
                                                            (short)0, acc[1][t], false, false);
      }
    }
  };

  ASet sA, sB;
  stageB(0, 0);
  loadA(0, sA);

  for (int s = 0; s < ns; s += 2) {
    // ---- even stage: buffer 0 / set A ----
    if (s + 1 < ns) { stageB(s + 1, 1); WAIT_ASYNC_LE(4); }
    else            { WAIT_ASYNC_LE(0); }
    __syncthreads();
    if (s + 1 < ns) loadA(s + 1, sB);
    compute(0, sA);
    __syncthreads();
    if (s + 1 < ns) {
      // ---- odd stage: buffer 1 / set B ----
      if (s + 2 < ns) { stageB(s + 2, 0); WAIT_ASYNC_LE(4); }
      else            { WAIT_ASYNC_LE(0); }
      __syncthreads();
      if (s + 2 < ns) loadA(s + 2, sA);
      compute(1, sB);
      __syncthreads();
    }
  }

  // Epilogue. C/D layout: VGPR r holds element (m = r + 8*half, n = lane&15).
  #pragma unroll
  for (int mt = 0; mt < 2; ++mt) {
    #pragma unroll
    for (int t = 0; t < 4; ++t) {
      const int n0 = nb + t * 16 + lm;
      const float bi = bias[n0];
      #pragma unroll
      for (int r = 0; r < 8; ++r) {
        const int m = mb + mt * 16 + r + 8 * half;
        const float vv = acc[mt][t][r] + bi;
        if (mode == 3) {
          outF[(size_t)m * NCOLS + n0] = vv;
        } else {
          const int b  = m >> 11;           // / SEQ
          const int s  = m & (SEQ - 1);
          const int hh = n0 >> 9;           // / DH
          const int dh = n0 & (DH - 1);
          size_t idx;
          if (mode == 2)  // V transposed: [B,H,DH,S]
            idx = (((size_t)(b * NH + hh)) * DH + dh) * SEQ + s;
          else            // Q/K: [B,H,S,DH]
            idx = (((size_t)(b * NH + hh)) * SEQ + s) * DH + dh;
          outB[idx] = f2bf(vv);
        }
      }
    }
  }
}

// ---------------------------------------------------------------------------
// Attention helper: exponentiate one 2048-wide bf16 row in LDS, return row sum
// ---------------------------------------------------------------------------
__device__ __forceinline__ float row_exp_pass(unsigned short* rowp, float rm, int lane) {
  float part = 0.f;
  for (int ch = 0; ch < 8; ++ch) {
    uint4 d = *(const uint4*)(rowp + ch * 256 + lane * 8);
    unsigned int w[4] = {d.x, d.y, d.z, d.w};
    unsigned int o[4];
    #pragma unroll
    for (int j = 0; j < 4; ++j) {
      float e0 = __expf(bf2f((unsigned short)(w[j] & 0xffffu)) - rm);
      float e1 = __expf(bf2f((unsigned short)(w[j] >> 16)) - rm);
      part += e0 + e1;
      o[j] = (unsigned int)f2bf(e0) | ((unsigned int)f2bf(e1) << 16);
    }
    uint4 dd; dd.x = o[0]; dd.y = o[1]; dd.z = o[2]; dd.w = o[3];
    *(uint4*)(rowp + ch * 256 + lane * 8) = dd;
  }
  part += __shfl_xor(part, 1, 32);
  part += __shfl_xor(part, 2, 32);
  part += __shfl_xor(part, 4, 32);
  part += __shfl_xor(part, 8, 32);
  part += __shfl_xor(part, 16, 32);
  return part;
}

// ---------------------------------------------------------------------------
// Kernel 4: attention. One wave per (b,h,16-query tile). K tiles async-staged
// into double-buffered LDS; full score row in padded LDS (bf16); two-pass
// softmax; un-normalized P*V with 1/rowsum folded into the store.
// Mask reproduces the reference: key <= query gets -1e9.
// ---------------------------------------------------------------------------
__global__ __launch_bounds__(32) void attn_kernel(const unsigned short* __restrict__ Q,
                                                  const unsigned short* __restrict__ Kt,
                                                  const unsigned short* __restrict__ Vt,
                                                  unsigned short* __restrict__ attn_out) {
  __shared__ unsigned short P[16 * PSTRIDE];        // 65792B
  __shared__ unsigned short Kst[2][16 * KSTRIDE];   // 2 x 16640B

  const int tile = (int)blockIdx.x;
  const int qt   = tile & (SEQ / 16 - 1);
  const int bh   = tile >> 7;
  const int b    = bh / NH;
  const int hh   = bh % NH;
  const int lane = threadIdx.x;
  const int lm   = lane & 15;
  const int half = lane >> 4;

  const unsigned short* Qb = Q  + (size_t)bh * SEQ * DH;
  const unsigned short* Kb = Kt + (size_t)bh * SEQ * DH;
  const unsigned short* Vb = Vt + (size_t)bh * DH * SEQ;
  const unsigned kstBase = LDS_OFF(&Kst[0][0]);

  auto stageK = [&](int kt, int buf) {
    const char* g0 = (const char*)(Kb + (size_t)kt * 16 * DH);
    const unsigned base = kstBase + (unsigned)buf * (16 * KSTRIDE * 2);
    #pragma unroll
    for (int i = 0; i < 32; ++i) {
      const int s = lane + i * 32;          // 0..1023 16B segments
      const int row = s >> 6, part = s & 63;
      async_b128(base + (unsigned)(row * (KSTRIDE * 2) + part * 16),
                 g0 + row * 1024 + part * 16);
    }
  };

  // Q row-tile resident in registers (A-operand layout)
  ABf qreg[16];
  const unsigned short* qrow = Qb + (size_t)(qt * 16 + lm) * DH;
  #pragma unroll
  for (int c = 0; c < 16; ++c) {
    qreg[c].h[0] = *(const v8bf*)(qrow + c * 32 + half * 16);
    qreg[c].h[1] = *(const v8bf*)(qrow + c * 32 + half * 16 + 8);
  }

  float rmax[8];
  #pragma unroll
  for (int r = 0; r < 8; ++r) rmax[r] = -3.0e38f;

  const float scale = 0.044194173824159216f;   // 1/sqrt(512)

  // ---- scores: 128 key tiles x 16 WMMAs, K streamed through LDS ----
  stageK(0, 0);
  for (int kt = 0; kt < SEQ / 16; ++kt) {
    const int buf = kt & 1;
    if (kt + 1 < SEQ / 16) {
      stageK(kt + 1, buf ^ 1);
      WAIT_ASYNC_LE(32);
    } else {
      WAIT_ASYNC_LE(0);
    }
    const unsigned short* krow = &Kst[buf][lm * KSTRIDE + half * 16];
    v8f acc = {};
    #pragma unroll
    for (int g = 0; g < 4; ++g) {          // groups of 4: batch B loads
      v16bf bm[4];
      #pragma unroll
      for (int j = 0; j < 4; ++j)
        bm[j] = *(const v16bf*)(krow + (g * 4 + j) * 32);
      #pragma unroll
      for (int j = 0; j < 4; ++j)
        acc = __builtin_amdgcn_wmma_f32_16x16x32_bf16(false, qreg[g * 4 + j].v, false, bm[j],
                                                      (short)0, acc, false, false);
    }
    const int key = kt * 16 + lm;
    #pragma unroll
    for (int r = 0; r < 8; ++r) {
      const int qpos = qt * 16 + r + 8 * half;
      float s = acc[r] * scale + ((key <= qpos) ? -1.0e9f : 0.0f);
      rmax[r] = fmaxf(rmax[r], s);
      P[(r + 8 * half) * PSTRIDE + key] = f2bf(s);
    }
  }

  #pragma unroll
  for (int r = 0; r < 8; ++r) {
    float m = rmax[r];
    m = fmaxf(m, __shfl_xor(m, 1, 32));
    m = fmaxf(m, __shfl_xor(m, 2, 32));
    m = fmaxf(m, __shfl_xor(m, 4, 32));
    m = fmaxf(m, __shfl_xor(m, 8, 32));
    rmax[r] = m;
  }
  __syncthreads();

  float inv[8];
  #pragma unroll
  for (int r = 0; r < 8; ++r) {
    const float rm0 = __shfl(rmax[r], 0, 32);
    const float rm1 = __shfl(rmax[r], 16, 32);
    const float p0 = row_exp_pass(&P[(r)     * PSTRIDE], rm0, lane);
    const float p1 = row_exp_pass(&P[(r + 8) * PSTRIDE], rm1, lane);
    inv[r] = 1.0f / (half ? p1 : p0);
  }
  __syncthreads();

  // ---- P * V: 32 dv tiles x 64 WMMAs; operands batch-loaded in groups of 4 ----
  for (int nt = 0; nt < DH / 16; ++nt) {
    const unsigned short* vrow = Vb + (size_t)(nt * 16 + lm) * SEQ + half * 16;
    __builtin_prefetch(vrow + 16 * SEQ, 0, 3);
    v8f acc = {};
    for (int cg = 0; cg < SEQ / 128; ++cg) {   // 16 groups of 4 chunks
      ABf a[4]; v16bf vv[4];
      #pragma unroll
      for (int j = 0; j < 4; ++j) {
        const int c = cg * 4 + j;
        a[j].h[0] = *(const v8bf*)(&P[lm * PSTRIDE + c * 32 + half * 16]);
        a[j].h[1] = *(const v8bf*)(&P[lm * PSTRIDE + c * 32 + half * 16 + 8]);
        vv[j] = *(const v16bf*)(vrow + c * 32);
      }
      #pragma unroll
      for (int j = 0; j < 4; ++j)
        acc = __builtin_amdgcn_wmma_f32_16x16x32_bf16(false, a[j].v, false, vv[j],
                                                      (short)0, acc, false, false);
    }
    const size_t colbase = (size_t)hh * DH + nt * 16 + lm;
    #pragma unroll
    for (int r = 0; r < 8; ++r) {
      const int m = r + 8 * half;
      const float o = acc[r] * inv[r];
      attn_out[(size_t)(b * SEQ + qt * 16 + m) * NCOLS + colbase] = f2bf(o);
    }
  }
}

// ---------------------------------------------------------------------------
// Host launch
// ---------------------------------------------------------------------------
extern "C" void kernel_launch(void* const* d_in, const int* in_sizes, int n_in,
                              void* d_out, int out_size, void* d_ws, size_t ws_size,
                              hipStream_t stream) {
  const float* x     = (const float*)d_in[0];
  const float* Wq    = (const float*)d_in[1];
  const float* bq    = (const float*)d_in[2];
  const float* Wk    = (const float*)d_in[3];
  const float* bk    = (const float*)d_in[4];
  const float* Wv    = (const float*)d_in[5];
  const float* bv    = (const float*)d_in[6];
  const float* Wo    = (const float*)d_in[7];
  const float* bo    = (const float*)d_in[8];
  const float* gamma = (const float*)d_in[9];
  const float* beta  = (const float*)d_in[10];
  float* out = (float*)d_out;

  char* ws = (char*)d_ws;
  size_t off = 0;
  unsigned short* xn  = (unsigned short*)(ws + off); off += (size_t)MROWS * DMODEL * 2;
  unsigned short* WqT = (unsigned short*)(ws + off); off += (size_t)NCOLS * DMODEL * 2;
  unsigned short* WkT = (unsigned short*)(ws + off); off += (size_t)NCOLS * DMODEL * 2;
  unsigned short* WvT = (unsigned short*)(ws + off); off += (size_t)NCOLS * DMODEL * 2;
  unsigned short* WoT = (unsigned short*)(ws + off); off += (size_t)NCOLS * NCOLS * 2;
  unsigned short* Qd  = (unsigned short*)(ws + off); off += (size_t)BATCH * NH * SEQ * DH * 2;
  unsigned short* Kd  = (unsigned short*)(ws + off); off += (size_t)BATCH * NH * SEQ * DH * 2;
  unsigned short* Vtd = (unsigned short*)(ws + off); off += (size_t)BATCH * NH * DH * SEQ * 2;
  unsigned short* At  = (unsigned short*)(ws + off); off += (size_t)MROWS * NCOLS * 2;

  prep_kernel<<<MROWS, 256, 0, stream>>>(x, gamma, beta, xn);

  dim3 tgQ(NCOLS / 32, DMODEL / 32);
  transpose_bf16<<<tgQ, 256, 0, stream>>>(Wq, WqT, DMODEL, NCOLS);
  transpose_bf16<<<tgQ, 256, 0, stream>>>(Wk, WkT, DMODEL, NCOLS);
  transpose_bf16<<<tgQ, 256, 0, stream>>>(Wv, WvT, DMODEL, NCOLS);
  dim3 tgO(NCOLS / 32, NCOLS / 32);
  transpose_bf16<<<tgO, 256, 0, stream>>>(Wo, WoT, NCOLS, NCOLS);

  dim3 gg(NCOLS / 64, MROWS / 128);
  gemm_bf16<<<gg, 128, 0, stream>>>(xn, WqT, bq, nullptr, Qd,  DMODEL, 0);
  gemm_bf16<<<gg, 128, 0, stream>>>(xn, WkT, bk, nullptr, Kd,  DMODEL, 0);
  gemm_bf16<<<gg, 128, 0, stream>>>(xn, WvT, bv, nullptr, Vtd, DMODEL, 2);

  attn_kernel<<<BATCH * NH * (SEQ / 16), 32, 0, stream>>>(Qd, Kd, Vtd, At);

  gemm_bf16<<<gg, 128, 0, stream>>>(At, WoT, bo, out, nullptr, NCOLS, 3);
}
